// MultiViewMatching_53343493816628
// MI455X (gfx1250) — matrix operand order repaired and checked by reading, not compiled, not run on hardware
//
#include <hip/hip_runtime.h>

// Problem constants (match reference)
#define NI 8192   // images
#define NC 8192   // captions
#define RV 7      // views per image
#define RP 8      // padded views per image
#define DK 256    // embedding dim

// LDS row stride in halves: 256 data + 8 pad = 264 (528 B).
// Makes both A-frag and B-frag ds_load_b128 patterns hit each bank exactly
// twice per instruction (the b128 minimum) -> conflict-free.
#define LSTR 264

typedef _Float16 h4   __attribute__((ext_vector_type(4)));
typedef _Float16 v8h  __attribute__((ext_vector_type(8)));
typedef _Float16 v16h __attribute__((ext_vector_type(16)));
typedef float    v8f  __attribute__((ext_vector_type(8)));

// ---------------------------------------------------------------------------
// Pre-pass 1: caps f32 -> f16, layout [NC][DK]
// ---------------------------------------------------------------------------
__global__ __launch_bounds__(256) void cvt_caps(const float* __restrict__ caps,
                                                _Float16* __restrict__ capsH) {
    int t = blockIdx.x * blockDim.x + threadIdx.x;   // one thread per 4 elems
    float4 f = ((const float4*)caps)[t];
    h4 o = { (_Float16)f.x, (_Float16)f.y, (_Float16)f.z, (_Float16)f.w };
    ((h4*)capsH)[t] = o;
}

// ---------------------------------------------------------------------------
// Pre-pass 2: imgs f32 [NI][7][DK] -> f16 padded [NI][8][DK], view 7 = zeros.
// A 16-row WMMA tile then covers exactly 2 consecutive padded images and the
// view-max becomes a pure per-lane reduction over acc[0..6].
// ---------------------------------------------------------------------------
__global__ __launch_bounds__(256) void pad_imgs(const float* __restrict__ imgs,
                                                _Float16* __restrict__ imgsH) {
    int t = blockIdx.x * blockDim.x + threadIdx.x;   // one thread per 4 elems
    int e   = t * 4;                  // element index in padded layout
    int img = e >> 11;                // / (8*256)
    int rem = e & 2047;
    int r   = rem >> 8;
    int k   = rem & 255;
    h4 o;
    if (r < RV) {
        const float4 f = *(const float4*)(imgs + ((size_t)img * RV + r) * DK + k);
        o = (h4){ (_Float16)f.x, (_Float16)f.y, (_Float16)f.z, (_Float16)f.w };
    } else {
        o = (h4){ (_Float16)0.f, (_Float16)0.f, (_Float16)0.f, (_Float16)0.f };
    }
    *(h4*)(imgsH + (size_t)e) = o;
}

// ---------------------------------------------------------------------------
// CDNA5 async global->LDS copy (ASYNCcnt path, no VGPR round-trip).
// vdst = LDS byte address, vaddr = 64-bit global address, saddr = off.
// ---------------------------------------------------------------------------
__device__ __forceinline__ void async_copy16(unsigned ldsAddr, const void* gptr) {
    asm volatile("global_load_async_to_lds_b128 %0, %1, off"
                 :: "v"(ldsAddr), "v"(gptr)
                 : "memory");
}

__device__ __forceinline__ void wait_async_all() {
    asm volatile("s_wait_asynccnt 0x0" ::: "memory");
}

// A fragment (16x32 f16): lane l holds row M=l%16; K = half*8 + {0..7,16..23}
__device__ __forceinline__ v16h lds_a_frag(const _Float16* p) {
    v8h lo = *(const v8h*)p;          // K = base .. base+7
    v8h hi = *(const v8h*)(p + 16);   // K = base+16 .. base+23
    v16h r;
#pragma unroll
    for (int i = 0; i < 8; ++i) { r[i] = lo[i]; r[i + 8] = hi[i]; }
    return r;
}

// B fragment (32x16 f16): lane l holds col N=l%16; K = half*8 + {0..7,16..23}
__device__ __forceinline__ v16h lds_b_frag(const _Float16* p) {
    v8h lo = *(const v8h*)p;
    v8h hi = *(const v8h*)(p + 16);
    v16h r;
#pragma unroll
    for (int i = 0; i < 8; ++i) { r[i] = lo[i]; r[i + 8] = hi[i]; }
    return r;
}

__device__ __forceinline__ float max7(v8f d) {
    float m = d[0];
    m = fmaxf(m, d[1]); m = fmaxf(m, d[2]); m = fmaxf(m, d[3]);
    m = fmaxf(m, d[4]); m = fmaxf(m, d[5]); m = fmaxf(m, d[6]);
    return m;   // element 7 is the zero-pad view -> excluded
}

// ---------------------------------------------------------------------------
// Main GEMM + view-max.
// Block = 256 threads = 8 waves arranged 4(M) x 2(N).
// Block tile: M = 128 rows (16 padded images), N = 128 caps, full K = 256.
// Wave tile:  M = 32 rows (2 M-subtiles), N = 64 (4 N-subtiles)
//             -> 8 WMMAs per K-step from 2 A frags + 4 B frags.
// LDS: A 128x264 + B 128x264 halves = 132 KB -> 2 blocks / WGP.
// ---------------------------------------------------------------------------
__global__ __launch_bounds__(256) void mvm_gemm(const _Float16* __restrict__ A,  // [NI][8][DK]
                                                const _Float16* __restrict__ B,  // [NC][DK]
                                                float* __restrict__ out) {       // [NI][NC]
    __shared__ _Float16 sA[128 * LSTR];
    __shared__ _Float16 sB[128 * LSTR];

    const int lane = threadIdx.x & 31;
    const int wave = threadIdx.x >> 5;
    const int half = lane >> 4;        // 0/1: image select within pair
    const int l16  = lane & 15;
    const int wm   = wave >> 1;        // 0..3 : M-wave
    const int wn   = wave & 1;         // 0..1 : N-wave

    // ---- Stage block tile into LDS with async copies --------------------
    {
        const _Float16* gA = A + (size_t)blockIdx.y * 128 * DK;
        const _Float16* gB = B + (size_t)blockIdx.x * 128 * DK;
        unsigned ldsA = (unsigned)(unsigned long long)(void*)sA;
        unsigned ldsB = (unsigned)(unsigned long long)(void*)sB;
#pragma unroll
        for (int i = 0; i < 16; ++i) {
            int flat = (int)threadIdx.x + i * 256;   // 0..4095 16B-chunks
            int row  = flat >> 5;                    // 32 chunks per 512B row
            int kb   = flat & 31;
            unsigned loff = (unsigned)(row * (LSTR * 2) + kb * 16);
            async_copy16(ldsA + loff, gA + (size_t)flat * 8);
            async_copy16(ldsB + loff, gB + (size_t)flat * 8);
        }
        wait_async_all();
        __syncthreads();
    }

    // ---- Register-blocked WMMA over K -----------------------------------
    const int aRow0 = wm * 32 + l16;          // M-subtile 0 row for this lane
    const int aRow1 = aRow0 + 16;             // M-subtile 1
    const int bCol  = wn * 64 + l16;          // N-subtile 0 col for this lane
    const int koffA = half * 8;
    const int koffB = half * 8;

    v8f acc[2][4] = {};

#pragma unroll
    for (int kk = 0; kk < DK; kk += 32) {
        v16h a0 = lds_a_frag(&sA[aRow0 * LSTR + kk + koffA]);
        v16h a1 = lds_a_frag(&sA[aRow1 * LSTR + kk + koffA]);
        v16h b0 = lds_b_frag(&sB[(bCol     ) * LSTR + kk + koffB]);
        v16h b1 = lds_b_frag(&sB[(bCol + 16) * LSTR + kk + koffB]);
        v16h b2 = lds_b_frag(&sB[(bCol + 32) * LSTR + kk + koffB]);
        v16h b3 = lds_b_frag(&sB[(bCol + 48) * LSTR + kk + koffB]);

        acc[0][0] = __builtin_amdgcn_wmma_f32_16x16x32_f16(false, a0, false, b0,
                                                           (short)0, acc[0][0], false, false);
        acc[0][1] = __builtin_amdgcn_wmma_f32_16x16x32_f16(false, a0, false, b1,
                                                           (short)0, acc[0][1], false, false);
        acc[0][2] = __builtin_amdgcn_wmma_f32_16x16x32_f16(false, a0, false, b2,
                                                           (short)0, acc[0][2], false, false);
        acc[0][3] = __builtin_amdgcn_wmma_f32_16x16x32_f16(false, a0, false, b3,
                                                           (short)0, acc[0][3], false, false);
        acc[1][0] = __builtin_amdgcn_wmma_f32_16x16x32_f16(false, a1, false, b0,
                                                           (short)0, acc[1][0], false, false);
        acc[1][1] = __builtin_amdgcn_wmma_f32_16x16x32_f16(false, a1, false, b1,
                                                           (short)0, acc[1][1], false, false);
        acc[1][2] = __builtin_amdgcn_wmma_f32_16x16x32_f16(false, a1, false, b2,
                                                           (short)0, acc[1][2], false, false);
        acc[1][3] = __builtin_amdgcn_wmma_f32_16x16x32_f16(false, a1, false, b3,
                                                           (short)0, acc[1][3], false, false);
    }

    // ---- Per-lane max over 7 valid views, coalesced nontemporal stores --
    const int imgBase = (int)blockIdx.y * 16 + wm * 4;   // 4 images per M-wave
    const int colBase = (int)blockIdx.x * 128 + wn * 64 + l16;
#pragma unroll
    for (int mi = 0; mi < 2; ++mi) {
        const size_t img = (size_t)(imgBase + mi * 2 + half);
        float* orow = out + img * (size_t)NC + colBase;
#pragma unroll
        for (int s = 0; s < 4; ++s) {
            __builtin_nontemporal_store(max7(acc[mi][s]), orow + s * 16);
        }
    }
}

// ---------------------------------------------------------------------------
// Launch. Workspace layout: capsH (4 MB) | imgsH (33.5 MB)  => needs ~37.6 MB.
// ---------------------------------------------------------------------------
extern "C" void kernel_launch(void* const* d_in, const int* in_sizes, int n_in,
                              void* d_out, int out_size, void* d_ws, size_t ws_size,
                              hipStream_t stream) {
    (void)in_sizes; (void)n_in; (void)out_size; (void)ws_size;

    const float* imgs = (const float*)d_in[0];   // [NI][RV][DK] f32
    const float* caps = (const float*)d_in[1];   // [NC][DK]     f32
    float*       out  = (float*)d_out;           // [NI][NC]     f32

    _Float16* capsH = (_Float16*)d_ws;                          // NC*DK halves
    _Float16* imgsH = capsH + (size_t)NC * DK;                  // NI*RP*DK halves

    {   // caps f32 -> f16
        int nvec = NC * DK / 4;
        cvt_caps<<<nvec / 256, 256, 0, stream>>>(caps, capsH);
    }
    {   // imgs f32 -> f16 padded to 8 views
        int nvec = NI * RP * DK / 4;
        pad_imgs<<<nvec / 256, 256, 0, stream>>>(imgs, imgsH);
    }
    {   // GEMM + max
        dim3 grid(NC / 128, NI / 16);
        mvm_gemm<<<grid, 256, 0, stream>>>(imgsH, capsH, out);
    }
}
